// DiffAttn_13417477833075
// MI455X (gfx1250) — compile-verified, matrix operand order checked by reading, not run on hardware
//
#include <hip/hip_runtime.h>

// ---------------------------------------------------------------------------
// DiffAttn on MI455X (gfx1250):
//  - all matmuls on v_wmma_f32_16x16x32_bf16 (f32 accumulate)
//  - S^T = K*Q^T formulation: softmax + P->PV handoff entirely in registers
//  - K / V^T blocks staged into LDS by the Tensor Data Mover (tensor_load_to_lds),
//    double-buffered on TENSORcnt, shared by the 4 waves of the workgroup,
//    with TDM row padding to kill LDS bank conflicts.
// ---------------------------------------------------------------------------

constexpr int D_   = 128;
constexpr int EMB_ = 1024;
constexpr int B_   = 4;
constexpr int S_   = 4096;
constexpr int TD_  = 256;   // 2*D

// padded LDS row strides (elements) produced by TDM pad fields
constexpr int KLD_STRIDE = 264;  // 256 + 8  (512B rows + 16B pad)
constexpr int VLD_STRIDE = 40;   // 32  + 8  (64B rows + 16B pad)

typedef __attribute__((ext_vector_type(16))) __bf16 v16bf;
typedef __attribute__((ext_vector_type(8)))  __bf16 v8bf;
typedef __attribute__((ext_vector_type(8)))  float  v8f;
typedef __attribute__((ext_vector_type(4)))  float  v4f;
typedef __attribute__((ext_vector_type(4)))  unsigned int u32x4;
typedef __attribute__((ext_vector_type(8)))  int    i32x8;
typedef __attribute__((ext_vector_type(4)))  int    i32x4;
typedef __bf16 bf16_t;

static __device__ __forceinline__ v8f wmma_bf16(v16bf a, v16bf b, v8f c) {
  return __builtin_amdgcn_wmma_f32_16x16x32_bf16(false, a, false, b,
                                                 (short)0, c, false, false);
}

// 16-bit A/B fragment from row-major memory (global or LDS). Lane L holds
// row L%16 with k in {kh..kh+7, 16+kh..16+kh+7}, kh = (L<16)?0:8.
static __device__ __forceinline__ v16bf frag_bf16(const bf16_t* p, int kh) {
  v8bf lo = *(const v8bf*)(p + kh);
  v8bf hi = *(const v8bf*)(p + 16 + kh);
  return __builtin_shufflevector(lo, hi, 0,1,2,3,4,5,6,7,8,9,10,11,12,13,14,15);
}

static __device__ __forceinline__ v16bf frag_f32(const float* p, int kh) {
  v4f a0 = *(const v4f*)(p + kh);
  v4f a1 = *(const v4f*)(p + kh + 4);
  v4f b0 = *(const v4f*)(p + 16 + kh);
  v4f b1 = *(const v4f*)(p + 16 + kh + 4);
  v16bf r;
#pragma unroll
  for (int i = 0; i < 4; ++i) {
    r[i]      = (__bf16)a0[i];
    r[4 + i]  = (__bf16)a1[i];
    r[8 + i]  = (__bf16)b0[i];
    r[12 + i] = (__bf16)b1[i];
  }
  return r;
}

static __device__ __forceinline__ float vmax8(v8f v) {
  float m = v[0];
#pragma unroll
  for (int i = 1; i < 8; ++i) m = fmaxf(m, v[i]);
  return m;
}
static __device__ __forceinline__ float vsum8(v8f v) {
  float s = v[0];
#pragma unroll
  for (int i = 1; i < 8; ++i) s += v[i];
  return s;
}

// Online softmax over a 32-key block held as two S^T C-tiles.
// Lane = query; lane pair (L, L+16) holds the query's 32 key values in exactly
// the per-lane positions the next WMMA's 16-bit B operand requires.
static __device__ __forceinline__ v16bf online_softmax(v8f s0, v8f s1,
                                                       float& m, float& l,
                                                       float& alpha) {
  float pm = fmaxf(vmax8(s0), vmax8(s1));
  pm = fmaxf(pm, __shfl_xor(pm, 16, 32));
  float mn = fmaxf(m, pm);
  alpha = __expf(m - mn);
  v8f p0, p1;
#pragma unroll
  for (int i = 0; i < 8; ++i) {
    p0[i] = __expf(s0[i] - mn);
    p1[i] = __expf(s1[i] - mn);
  }
  float ps = vsum8(p0) + vsum8(p1);
  ps += __shfl_xor(ps, 16, 32);
  l = l * alpha + ps;
  m = mn;
  v16bf bp;
#pragma unroll
  for (int i = 0; i < 8; ++i) {
    bp[i]     = (__bf16)p0[i];
    bp[8 + i] = (__bf16)p1[i];
  }
  return bp;
}

// ------------------------- TDM: 2-D tile -> LDS ----------------------------
// D# per CDNA5 ISA. data_size=2B, tensor==tile
// (tiles always in-bounds), optional LDS row padding via pad fields.
static __device__ __forceinline__ uint32_t lds_byte_off(const bf16_t* p) {
  return (uint32_t)(uintptr_t)(const __attribute__((address_space(3))) bf16_t*)p;
}

static __device__ __forceinline__ void tdm_load_2d(
    const bf16_t* lds_dst, const bf16_t* gsrc,
    uint32_t tile_d0, uint32_t tile_d1, uint64_t stride0_elems,
    uint32_t pad_interval_code, uint32_t pad_amount_code) {
  const uint64_t ga = (uint64_t)(uintptr_t)gsrc;
  u32x4 g0;
  g0[0] = 1u;                                   // count=1, user descriptor
  g0[1] = lds_byte_off(lds_dst);                // lds_addr
  g0[2] = (uint32_t)(ga & 0xFFFFFFFFu);         // global_addr[31:0]
  g0[3] = (uint32_t)((ga >> 32) & 0x01FFFFFFu)  // global_addr[56:32]
          | (2u << 30);                         // type = 2 ("image")
  uint32_t w0 = (1u << 16);                     // data_size = 1 -> 2 bytes
  if (pad_amount_code)
    w0 |= (1u << 20)                            // pad_enable
        | (pad_interval_code << 22)             // pad_interval
        | (pad_amount_code << 25);              // pad_amount
  i32x8 g1;
  g1[0] = (int)w0;
  g1[1] = (int)((tile_d0 & 0xFFFFu) << 16);                      // tensor_dim0 lo
  g1[2] = (int)((tile_d0 >> 16) | ((tile_d1 & 0xFFFFu) << 16));  // td0 hi | td1 lo
  g1[3] = (int)((tile_d1 >> 16) | (tile_d0 << 16));              // td1 hi | tile_dim0
  g1[4] = (int)(tile_d1 & 0xFFFFu);                              // tile_dim1 | tile_dim2=0
  g1[5] = (int)(uint32_t)(stride0_elems & 0xFFFFFFFFu);          // dim0 stride lo
  g1[6] = (int)(uint32_t)((stride0_elems >> 32) & 0xFFFFu);      // dim0 stride hi
  g1[7] = 0;
  i32x4 z4 = {0, 0, 0, 0};
#if defined(__clang_major__) && (__clang_major__ >= 23)
  i32x8 z8 = {0, 0, 0, 0, 0, 0, 0, 0};
  __builtin_amdgcn_tensor_load_to_lds(g0, g1, z4, z4, z8, 0);
#else
  __builtin_amdgcn_tensor_load_to_lds(g0, g1, z4, z4, 0);
#endif
}

// --------------------------- kernel 1: W -> W^T bf16 -----------------------
__global__ void wt_kernel(const float* __restrict__ W, bf16_t* __restrict__ Wt,
                          int N, int total) {
  int idx = blockIdx.x * blockDim.x + threadIdx.x;
  if (idx >= total) return;
  int n = idx >> 10;
  int k = idx & (EMB_ - 1);
  Wt[(size_t)n * EMB_ + k] = (bf16_t)W[(size_t)k * N + n];
}

// ------------------- kernel 2: projection GEMM (X @ W + b) -----------------
__global__ __launch_bounds__(128) void proj_kernel(
    const float* __restrict__ X, const bf16_t* __restrict__ Wt,
    const float* __restrict__ bias, bf16_t* __restrict__ out,
    int N, int vtrans) {
  const int lane = threadIdx.x & 31;
  const int wid  = (blockIdx.x * blockDim.x + threadIdx.x) >> 5;
  const int nstrips = N >> 6;
  const int mt = wid / nstrips;
  const int ns = wid % nstrips;
  const int m0 = mt * 16;
  const int row = lane & 15;
  const int kh  = (lane < 16) ? 0 : 8;
  const int moff = (lane < 16) ? 0 : 8;

  v8f acc[4] = {};
  const float* xrow = X + (size_t)(m0 + row) * EMB_;
  for (int kc = 0; kc < EMB_; kc += 32) {
    v16bf a = frag_f32(xrow + kc, kh);
#pragma unroll
    for (int nt = 0; nt < 4; ++nt) {
      const bf16_t* wrow = Wt + (size_t)(ns * 64 + nt * 16 + row) * EMB_ + kc;
      acc[nt] = wmma_bf16(a, frag_bf16(wrow, kh), acc[nt]);
    }
  }
#pragma unroll
  for (int nt = 0; nt < 4; ++nt) {
    const int n = ns * 64 + nt * 16 + row;
    const float bv = bias[n];
    if (!vtrans) {
#pragma unroll
      for (int r = 0; r < 8; ++r)
        out[(size_t)(m0 + moff + r) * N + n] = (bf16_t)(acc[nt][r] + bv);
    } else {
      const int bidx = m0 / S_;
      const int s0   = (m0 & (S_ - 1)) + moff;
      bf16_t* dst = out + ((size_t)(bidx * D_ + n)) * S_ + s0;
#pragma unroll
      for (int r = 0; r < 8; ++r)
        dst[r] = (bf16_t)(acc[nt][r] + bv);
    }
  }
}

// ----------------- kernel 3: dual-stream flash attention -------------------
// 4 waves / block = 4 query tiles of one batch. K and V^T blocks are staged
// into LDS by the TDM (double-buffered, padded rows), shared by all waves.
__global__ __launch_bounds__(128) void attn_kernel(
    const bf16_t* __restrict__ Qb, const bf16_t* __restrict__ Kb,
    const bf16_t* __restrict__ Vt, const float* __restrict__ lam_p,
    float* __restrict__ out) {
  __shared__ bf16_t Klds[2][32 * KLD_STRIDE];    // 2 x 16.5 KB
  __shared__ bf16_t Vlds[2][128 * VLD_STRIDE];   // 2 x 10.0 KB

  const int lane = threadIdx.x & 31;
  const int warp = threadIdx.x >> 5;
  const int b    = blockIdx.x >> 6;              // 64 blocks per batch
  const int qt   = (blockIdx.x & 63) * 4 + warp;
  const int row  = lane & 15;
  const int kh   = (lane < 16) ? 0 : 8;
  const float lam = *lam_p;
  const float sscale = 0.088388347648318447f;    // 1/sqrt(128)

  // Q fragments (B operands of K*Q^T): stream1 cols 0..127, stream2 128..255
  const bf16_t* qrow = Qb + ((size_t)(b * S_ + qt * 16 + row)) * TD_;
  v16bf q1[4], q2[4];
#pragma unroll
  for (int kc = 0; kc < 4; ++kc) {
    q1[kc] = frag_bf16(qrow + kc * 32, kh);
    q2[kc] = frag_bf16(qrow + D_ + kc * 32, kh);
  }

  const bf16_t* kblk = Kb + (size_t)b * S_ * TD_;        // + j*TD_ per block
  const bf16_t* vblk = Vt + (size_t)b * D_ * S_;         // + j per block

  // prologue: DMA block j=0 into buffer 0 (wave 0 only; TDM ignores EXEC)
  if (warp == 0) {
    tdm_load_2d(&Klds[0][0], kblk, TD_, 32, TD_, 6, 3);  // 128DW rows, +4DW pad
    tdm_load_2d(&Vlds[0][0], vblk, 32, D_, S_, 3, 3);    // 16DW rows,  +4DW pad
  }

  v8f o1[8] = {};
  v8f o2[8] = {};
  float m1 = -1e30f, l1 = 0.0f, m2 = -1e30f, l2 = 0.0f;

  for (int j = 0; j < S_; j += 32) {
    const int cur = (j >> 5) & 1;
    const int nxt = cur ^ 1;

    if (warp == 0) __builtin_amdgcn_s_wait_tensorcnt(0);  // buf[cur] ready
    __syncthreads();                                      // publish to all waves
    if (warp == 0 && j + 32 < S_) {                       // overlap DMA(j+32)
      tdm_load_2d(&Klds[nxt][0], kblk + (size_t)(j + 32) * TD_, TD_, 32, TD_, 6, 3);
      tdm_load_2d(&Vlds[nxt][0], vblk + (j + 32), 32, D_, S_, 3, 3);
    }

    // S^T = K * Q^T for both streams (keys j..j+31)
    v8f s10 = {}, s11 = {}, s20 = {}, s21 = {};
    const bf16_t* kr0 = &Klds[cur][(size_t)row * KLD_STRIDE];
    const bf16_t* kr1 = kr0 + 16 * KLD_STRIDE;
#pragma unroll
    for (int kc = 0; kc < 4; ++kc) {
      s10 = wmma_bf16(frag_bf16(kr0 + kc * 32, kh),      q1[kc], s10);
      s20 = wmma_bf16(frag_bf16(kr0 + D_ + kc * 32, kh), q2[kc], s20);
      s11 = wmma_bf16(frag_bf16(kr1 + kc * 32, kh),      q1[kc], s11);
      s21 = wmma_bf16(frag_bf16(kr1 + D_ + kc * 32, kh), q2[kc], s21);
    }
    s10 *= sscale; s11 *= sscale; s20 *= sscale; s21 *= sscale;

    float a1, a2;
    v16bf bp1 = online_softmax(s10, s11, m1, l1, a1);
    v16bf bp2 = online_softmax(s20, s21, m2, l2, a2);
#pragma unroll
    for (int i = 0; i < 8; ++i) { o1[i] *= a1; o2[i] *= a2; }

    // O^T += V^T * P^T
    const bf16_t* vr = &Vlds[cur][(size_t)row * VLD_STRIDE];
#pragma unroll
    for (int dt = 0; dt < 8; ++dt) {
      v16bf av = frag_bf16(vr + dt * 16 * VLD_STRIDE, kh);
      o1[dt] = wmma_bf16(av, bp1, o1[dt]);
      o2[dt] = wmma_bf16(av, bp2, o2[dt]);
    }
    __syncthreads();   // all waves done with buf[cur] before it is re-filled
  }

  // Epilogue: out[b][q][d] = o1/l1 - lam*o2/l2 ; lane = query, VGPR = d-row.
  const float i1 = 1.0f / l1;
  const float i2 = lam / l2;
  const int q    = qt * 16 + row;
  const int doff = (lane < 16) ? 0 : 8;
  float* orow = out + ((size_t)(b * S_ + q)) * D_;
#pragma unroll
  for (int dt = 0; dt < 8; ++dt) {
    v8f r = o1[dt] * i1 - o2[dt] * i2;
    float4 lo = make_float4(r[0], r[1], r[2], r[3]);
    float4 hi = make_float4(r[4], r[5], r[6], r[7]);
    *(float4*)(orow + dt * 16 + doff)     = lo;
    *(float4*)(orow + dt * 16 + doff + 4) = hi;
  }
}

// ---------------------------------------------------------------------------
extern "C" void kernel_launch(void* const* d_in, const int* in_sizes, int n_in,
                              void* d_out, int out_size, void* d_ws, size_t ws_size,
                              hipStream_t stream) {
  (void)in_sizes; (void)n_in; (void)out_size; (void)ws_size;
  const float* X   = (const float*)d_in[0];
  const float* lam = (const float*)d_in[1];
  const float* Wq  = (const float*)d_in[2];
  const float* bq  = (const float*)d_in[3];
  const float* Wk  = (const float*)d_in[4];
  const float* bk  = (const float*)d_in[5];
  const float* Wv  = (const float*)d_in[6];
  const float* bv  = (const float*)d_in[7];

  // Workspace (bf16): Qb 8MB | Kb 8MB | Vt 4MB | Wt's 1.3MB  (~21.3MB)
  bf16_t* Qb  = (bf16_t*)d_ws;
  bf16_t* Kbf = Qb  + (size_t)B_ * S_ * TD_;
  bf16_t* Vt  = Kbf + (size_t)B_ * S_ * TD_;
  bf16_t* Wtq = Vt  + (size_t)B_ * D_ * S_;
  bf16_t* Wtk = Wtq + (size_t)TD_ * EMB_;
  bf16_t* Wtv = Wtk + (size_t)TD_ * EMB_;

  wt_kernel<<<(TD_ * EMB_ + 255) / 256, 256, 0, stream>>>(Wq, Wtq, TD_, TD_ * EMB_);
  wt_kernel<<<(TD_ * EMB_ + 255) / 256, 256, 0, stream>>>(Wk, Wtk, TD_, TD_ * EMB_);
  wt_kernel<<<(D_  * EMB_ + 255) / 256, 256, 0, stream>>>(Wv, Wtv, D_, D_ * EMB_);

  proj_kernel<<<(B_ * S_ / 16) * (TD_ / 64) / 4, 128, 0, stream>>>(X, Wtq, bq, Qb,  TD_, 0);
  proj_kernel<<<(B_ * S_ / 16) * (TD_ / 64) / 4, 128, 0, stream>>>(X, Wtk, bk, Kbf, TD_, 0);
  proj_kernel<<<(B_ * S_ / 16) * (D_  / 64) / 4, 128, 0, stream>>>(X, Wtv, bv, Vt,  D_, 1);

  attn_kernel<<<(B_ * S_ / 16) / 4, 128, 0, stream>>>(Qb, Kbf, Vt, lam, (float*)d_out);
}
